// TransLayer_9869834847106
// MI455X (gfx1250) — matrix operand order repaired and checked by reading, not compiled, not run on hardware
//
#include <hip/hip_runtime.h>

// ---------------------------------------------------------------------------
// Nystrom-attention transformer layer for MI455X (gfx1250, wave32, WMMA).
// Heavy math: v_wmma_f32_16x16x32_bf16 (f32 accumulate).
// Data movement: global_load_async_to_lds_b128 + s_wait_asynccnt, double-
// buffered LDS, one barrier per K-step.
// Workspace requirement: ~480 MB (buffers aliased where lifetimes allow).
// ---------------------------------------------------------------------------

typedef __attribute__((ext_vector_type(16))) __bf16 v16bf;
typedef __attribute__((ext_vector_type(8)))  float  v8f;
typedef __attribute__((ext_vector_type(4)))  float  f32x4;

union FragU { v16bf v; f32x4 q[2]; };

// Problem constants
#define BB   8
#define NN   4096
#define DD   512
#define HH   8
#define DH   64
#define MM   256
#define BN   (BB*NN)      // 32768
#define BH   (BB*HH)      // 64
#define RES_K 33

// CDNA5 async copy: global -> LDS, 16B per lane, tracked by ASYNCcnt.
__device__ __forceinline__ void async_ld_b128(unsigned lds_off, const __bf16* g)
{
    asm volatile("global_load_async_to_lds_b128 %0, %1, off"
                 :: "v"(lds_off), "v"(g) : "memory");
}
__device__ __forceinline__ void wait_async0()
{
    asm volatile("s_wait_asynccnt 0x0" ::: "memory");
}
__device__ __forceinline__ unsigned lds_off_of(const void* p)
{
    // generic pointer to shared: low 32 bits are the wave-relative LDS offset
    return (unsigned)(unsigned long long)p;
}

// ---------------------------------------------------------------------------
// Generic bf16 WMMA GEMM:  C[M,N] = alpha * A[M,K] x B + diag * I  (+bias+res)
//   BT=true : B given transposed, Bt[N,K] row-major (ldb = row stride of Bt)
//   BT=false: B given [K,N] row-major (transposed while staging into LDS)
// Block = 128 threads (4 waves); block tile 64x64; K-step 32.
// Each wave owns a 32x32 region = 2x2 WMMA 16x16 tiles.
// ---------------------------------------------------------------------------
template<bool BT, bool OUT_BF16, bool BIAS_RES>
__global__ __launch_bounds__(128)
void k_gemm(const __bf16* __restrict__ A, const __bf16* __restrict__ B,
            void* __restrict__ Cv, int K, int lda, int ldb, int ldc,
            long sA, long sB, long sC, float alpha, float diag,
            const float* __restrict__ bias, const float* __restrict__ resid)
{
    const int bz = blockIdx.z;
    A += (size_t)bz * sA;
    B += (size_t)bz * sB;
    const int m0 = blockIdx.y << 6;
    const int n0 = blockIdx.x << 6;
    const int tid  = threadIdx.x;
    const int lane = tid & 31;
    const int wave = tid >> 5;
    const int L = lane & 15;          // row/col within 16x16 tile
    const int g = lane >> 4;          // half-wave group
    const int wr = (wave >> 1) << 5;  // wave row offset in block tile
    const int wc = (wave & 1) << 5;   // wave col offset in block tile

    __shared__ alignas(16) __bf16 As[2][64][40];  // double-buffered 64x32 (+pad)
    __shared__ alignas(16) __bf16 Bs[2][64][40];  // Bs[n][k]

    // staging coordinates (2 x 16B chunks per thread)
    const int ar0  = tid >> 2,         ak0 = (tid & 3) << 3;         // chunk 0
    const int ar1  = (tid + 128) >> 2, ak1 = ((tid + 128) & 3) << 3; // chunk 1
    const int bkr0 = tid >> 3,         bn0 = (tid & 7) << 3;         // non-BT
    const int bkr1 = (tid + 128) >> 3, bn1 = ((tid + 128) & 7) << 3;

    const v8f vzero = {0.f,0.f,0.f,0.f,0.f,0.f,0.f,0.f};
    v8f c00 = vzero, c01 = vzero, c10 = vzero, c11 = vzero;

    f32x4 breg0 = {}, breg1 = {};  // register pipeline for non-BT B tile

    auto issueA = [&](int kt, int buf) {
        const int k0 = kt << 5;
        async_ld_b128(lds_off_of(&As[buf][ar0][ak0]),
                      A + (size_t)(m0 + ar0) * lda + k0 + ak0);
        async_ld_b128(lds_off_of(&As[buf][ar1][ak1]),
                      A + (size_t)(m0 + ar1) * lda + k0 + ak1);
    };
    auto issueBt = [&](int kt, int buf) {
        const int k0 = kt << 5;
        async_ld_b128(lds_off_of(&Bs[buf][ar0][ak0]),
                      B + (size_t)(n0 + ar0) * ldb + k0 + ak0);
        async_ld_b128(lds_off_of(&Bs[buf][ar1][ak1]),
                      B + (size_t)(n0 + ar1) * ldb + k0 + ak1);
    };
    auto loadB = [&](int kt) {
        const int k0 = kt << 5;
        breg0 = *(const f32x4*)(B + (size_t)(k0 + bkr0) * ldb + n0 + bn0);
        breg1 = *(const f32x4*)(B + (size_t)(k0 + bkr1) * ldb + n0 + bn1);
    };
    auto storeB = [&](int buf) {
        union { f32x4 f; __bf16 h[8]; } u;
        u.f = breg0;
        #pragma unroll
        for (int e = 0; e < 8; ++e) Bs[buf][bn0 + e][bkr0] = u.h[e];
        u.f = breg1;
        #pragma unroll
        for (int e = 0; e < 8; ++e) Bs[buf][bn1 + e][bkr1] = u.h[e];
    };

    const int nk = K >> 5;
    // prologue: stage tile 0 into buffer 0
    issueA(0, 0);
    if constexpr (BT) { issueBt(0, 0); }
    else              { loadB(0); storeB(0); }

    for (int kt = 0; kt < nk; ++kt) {
        const int buf = kt & 1;
        wait_async0();         // my async writes for tile kt have landed
        __syncthreads();       // everyone's writes visible; prior reads done

        const bool more = (kt + 1 < nk);
        if (more) {
            // overlap next tile's copy with this tile's WMMAs
            issueA(kt + 1, buf ^ 1);
            if constexpr (BT) issueBt(kt + 1, buf ^ 1);
            else              loadB(kt + 1);
        }

        // ---- fragments per CDNA5 16-bit layouts ----
        // A(16x32): lane (L,g): elems 0..7 = K 8g.. ; elems 8..15 = K 16+8g..
        // B(32x16): lane (L,g): elems 0..7 = K 8g.. ; elems 8..15 = K 16+8g.. of column L
        FragU a0, a1, b0, b1;
        a0.q[0] = *(const f32x4*)&As[buf][wr + L][(g << 3)];
        a0.q[1] = *(const f32x4*)&As[buf][wr + L][16 + (g << 3)];
        a1.q[0] = *(const f32x4*)&As[buf][wr + 16 + L][(g << 3)];
        a1.q[1] = *(const f32x4*)&As[buf][wr + 16 + L][16 + (g << 3)];
        b0.q[0] = *(const f32x4*)&Bs[buf][wc + L][(g << 3)];
        b0.q[1] = *(const f32x4*)&Bs[buf][wc + L][16 + (g << 3)];
        b1.q[0] = *(const f32x4*)&Bs[buf][wc + 16 + L][(g << 3)];
        b1.q[1] = *(const f32x4*)&Bs[buf][wc + 16 + L][16 + (g << 3)];

        c00 = __builtin_amdgcn_wmma_f32_16x16x32_bf16(false, a0.v, false, b0.v, (short)0, c00, false, false);
        c01 = __builtin_amdgcn_wmma_f32_16x16x32_bf16(false, a0.v, false, b1.v, (short)0, c01, false, false);
        c10 = __builtin_amdgcn_wmma_f32_16x16x32_bf16(false, a1.v, false, b0.v, (short)0, c10, false, false);
        c11 = __builtin_amdgcn_wmma_f32_16x16x32_bf16(false, a1.v, false, b1.v, (short)0, c11, false, false);

        if constexpr (!BT) {
            if (more) storeB(buf ^ 1);  // loadcnt wait lands behind the WMMAs
        }
    }

    // ---- epilogue: C layout VGPR r -> row 8g+r, col L ----
    float*  Cf = (float*)Cv;
    __bf16* Cb = (__bf16*)Cv;
    v8f accs[2][2] = {{c00, c01}, {c10, c11}};
    #pragma unroll
    for (int ti = 0; ti < 2; ++ti) {
        #pragma unroll
        for (int tj = 0; tj < 2; ++tj) {
            const int gm_base = m0 + wr + (ti << 4) + (g << 3);
            const int gn      = n0 + wc + (tj << 4) + L;
            #pragma unroll
            for (int r = 0; r < 8; ++r) {
                const int gm = gm_base + r;
                float val = alpha * accs[ti][tj][r];
                if (gm == gn) val += diag;
                if (BIAS_RES) val += bias[gn] + resid[(size_t)gm * ldc + gn];
                const size_t off = (size_t)bz * sC + (size_t)gm * ldc + gn;
                if (OUT_BF16) Cb[off] = (__bf16)val;
                else          Cf[off] = val;
            }
        }
    }
}

// ---------------------------------------------------------------------------
// LayerNorm: one 256-thread block per row of 512; writes bf16.
// ---------------------------------------------------------------------------
__global__ __launch_bounds__(256)
void k_layernorm(const float* __restrict__ x, const float* __restrict__ gamma,
                 const float* __restrict__ beta, __bf16* __restrict__ xn)
{
    const int row = blockIdx.x, tid = threadIdx.x;
    const float* xr = x + (size_t)row * DD;
    float v0 = xr[tid], v1 = xr[tid + 256];
    __shared__ float s1[256], s2[256];
    s1[tid] = v0 + v1;
    s2[tid] = v0 * v0 + v1 * v1;
    __syncthreads();
    for (int o = 128; o > 0; o >>= 1) {
        if (tid < o) { s1[tid] += s1[tid + o]; s2[tid] += s2[tid + o]; }
        __syncthreads();
    }
    const float mu  = s1[0] * (1.f / DD);
    const float var = s2[0] * (1.f / DD) - mu * mu;
    const float rs  = rsqrtf(var + 1e-5f);
    xn[(size_t)row * DD + tid]       = (__bf16)((v0 - mu) * rs * gamma[tid] + beta[tid]);
    xn[(size_t)row * DD + tid + 256] = (__bf16)((v1 - mu) * rs * gamma[tid + 256] + beta[tid + 256]);
}

__global__ void k_cvt_bf16(const float* __restrict__ in, __bf16* __restrict__ out, int n)
{
    int i = blockIdx.x * 256 + threadIdx.x;
    if (i < n) out[i] = (__bf16)in[i];
}

// qkv_tmp[BN,1536] -> q,k,v [b,h,n,dh] bf16 (q pre-scaled by dh^-0.5 = 0.125)
__global__ void k_split_qkv(const __bf16* __restrict__ qkv, __bf16* __restrict__ q,
                            __bf16* __restrict__ k, __bf16* __restrict__ v)
{
    size_t idx = (size_t)blockIdx.x * 256 + threadIdx.x;  // over BN*512
    int row = (int)(idx >> 9), j = (int)(idx & 511);
    int bidx = row >> 12, pos = row & 4095;
    int h = j >> 6, d = j & 63;
    size_t dst = (((size_t)(bidx * HH + h) * NN) + pos) * DH + d;
    const __bf16* src = qkv + (size_t)row * 1536 + j;
    q[dst] = (__bf16)((float)src[0] * 0.125f);
    k[dst] = src[512];
    v[dst] = src[1024];
}

// landmark pooling: [bh,256*16,64] -> mean over 16 -> [bh,256,64]
__global__ void k_landmark(const __bf16* __restrict__ src, __bf16* __restrict__ dst)
{
    size_t idx = (size_t)blockIdx.x * 256 + threadIdx.x;  // BH*256*64
    int d  = (int)(idx & 63);
    int j  = (int)((idx >> 6) & 255);
    int bh = (int)(idx >> 14);
    const __bf16* p = src + ((size_t)bh * NN + (size_t)j * 16) * DH + d;
    float s = 0.f;
    #pragma unroll
    for (int i = 0; i < 16; ++i) s += (float)p[(size_t)i * DH];
    dst[((size_t)bh * MM + j) * DH + d] = (__bf16)(s * (1.f / 16.f));
}

// in-place row softmax on bf16, one wave per row
__global__ __launch_bounds__(256)
void k_softmax_bf16(__bf16* __restrict__ p, int width)
{
    size_t row = (size_t)blockIdx.x * 8 + (threadIdx.x >> 5);
    int lane = threadIdx.x & 31;
    __bf16* r = p + row * (size_t)width;
    float mx = -3.4e38f;
    for (int j = lane; j < width; j += 32) mx = fmaxf(mx, (float)r[j]);
    for (int o = 16; o; o >>= 1) mx = fmaxf(mx, __shfl_xor(mx, o, 32));
    float s = 0.f;
    for (int j = lane; j < width; j += 32) s += __expf((float)r[j] - mx);
    for (int o = 16; o; o >>= 1) s += __shfl_xor(s, o, 32);
    float inv = 1.f / s;
    for (int j = lane; j < width; j += 32) r[j] = (__bf16)(__expf((float)r[j] - mx) * inv);
}

// f32 in-place softmax (width 256) + bf16 mirror for WMMA consumption
__global__ __launch_bounds__(256)
void k_softmax_f32_dual(float* __restrict__ p, __bf16* __restrict__ pb)
{
    size_t row = (size_t)blockIdx.x * 8 + (threadIdx.x >> 5);
    int lane = threadIdx.x & 31;
    float*  r  = p  + row * 256;
    __bf16* rb = pb + row * 256;
    float mx = -3.4e38f;
    for (int j = lane; j < 256; j += 32) mx = fmaxf(mx, r[j]);
    for (int o = 16; o; o >>= 1) mx = fmaxf(mx, __shfl_xor(mx, o, 32));
    float s = 0.f;
    for (int j = lane; j < 256; j += 32) s += __expf(r[j] - mx);
    for (int o = 16; o; o >>= 1) s += __shfl_xor(s, o, 32);
    float inv = 1.f / s;
    for (int j = lane; j < 256; j += 32) {
        float e = __expf(r[j] - mx) * inv;
        r[j] = e; rb[j] = (__bf16)e;
    }
}

// per-bh max row-sum / max col-sum of attn2 (f32)
__global__ __launch_bounds__(256)
void k_pinv_norm(const float* __restrict__ a, float* __restrict__ nb)
{
    int bh = blockIdx.x, tid = threadIdx.x;
    const float* m = a + (size_t)bh * (MM * MM);
    float rs = 0.f, cs = 0.f;
    for (int j = 0; j < MM; ++j) {
        rs += fabsf(m[(size_t)tid * MM + j]);
        cs += fabsf(m[(size_t)j * MM + tid]);
    }
    __shared__ float shr[256], shc[256];
    shr[tid] = rs; shc[tid] = cs;
    __syncthreads();
    for (int o = 128; o; o >>= 1) {
        if (tid < o) { shr[tid] = fmaxf(shr[tid], shr[tid + o]);
                       shc[tid] = fmaxf(shc[tid], shc[tid + o]); }
        __syncthreads();
    }
    if (tid == 0) { nb[bh] = shr[0]; nb[64 + bh] = shc[0]; }
}

__global__ void k_pinv_scalar(float* __restrict__ nb)
{
    if (threadIdx.x == 0) {
        float mr = 0.f, mc = 0.f;
        for (int i = 0; i < 64; ++i) { mr = fmaxf(mr, nb[i]); mc = fmaxf(mc, nb[64 + i]); }
        nb[128] = 1.f / (mr * mc);
    }
}

// z0 = a^T * s  (bf16)
__global__ void k_z0(const float* __restrict__ a, const float* __restrict__ nb,
                     __bf16* __restrict__ z)
{
    size_t idx = (size_t)blockIdx.x * 256 + threadIdx.x;  // BH*65536
    float s = nb[128];
    int j = (int)(idx & 255), i = (int)((idx >> 8) & 255);
    size_t bh = idx >> 16;
    z[idx] = (__bf16)(a[(bh << 16) + ((size_t)j << 8) + i] * s);
}

// e = 7*I - az   (elementwise + diag, bf16)
__global__ void k_diag7(const __bf16* __restrict__ az, __bf16* __restrict__ e)
{
    size_t idx = (size_t)blockIdx.x * 256 + threadIdx.x;  // BH*65536
    int j = (int)(idx & 255), i = (int)((idx >> 8) & 255);
    e[idx] = (__bf16)(((i == j) ? 7.f : 0.f) - (float)az[idx]);
}

// depthwise 33-tap seq conv on v + add attention output -> concat [b,n,512] bf16
__global__ __launch_bounds__(256)
void k_conv_combine(const __bf16* __restrict__ v, const float* __restrict__ outh,
                    const float* __restrict__ rk, __bf16* __restrict__ concat)
{
    const int bh = blockIdx.y;               // 0..63
    const int b = bh >> 3, h = bh & 7;
    const int n0 = blockIdx.x << 8;          // 256 positions per block
    const int tid = threadIdx.x;
    __shared__ __bf16 lv[288 * 64];          // 16 halo each side
    __shared__ float  kw[RES_K];
    if (tid < RES_K) kw[tid] = rk[h * RES_K + tid];
    const __bf16* vb = v + (size_t)bh * NN * DH;
    for (int i = tid; i < 288 * 64; i += 256) {
        int p = i >> 6, d = i & 63;
        int gn = n0 - 16 + p;
        lv[i] = (gn >= 0 && gn < NN) ? vb[(size_t)gn * DH + d] : (__bf16)0.f;
    }
    __syncthreads();
    const float* ob = outh + (size_t)bh * NN * DH;
    for (int i = tid; i < 256 * 64; i += 256) {
        int p = i >> 6, d = i & 63;
        float acc = 0.f;
        #pragma unroll
        for (int t = 0; t < RES_K; ++t) acc += (float)lv[(p + t) * 64 + d] * kw[t];
        int gn = n0 + p;
        acc += ob[(size_t)gn * DH + d];
        concat[((size_t)b * NN + gn) * DD + h * DH + d] = (__bf16)acc;
    }
}

// ---------------------------------------------------------------------------
// Host orchestration
// ---------------------------------------------------------------------------
extern "C" void kernel_launch(void* const* d_in, const int* in_sizes, int n_in,
                              void* d_out, int out_size, void* d_ws, size_t ws_size,
                              hipStream_t stream)
{
    (void)in_sizes; (void)n_in; (void)out_size; (void)ws_size;
    const float* x     = (const float*)d_in[0];
    const float* gamma = (const float*)d_in[1];
    const float* beta  = (const float*)d_in[2];
    const float* w_qkv = (const float*)d_in[3];
    const float* w_out = (const float*)d_in[4];
    const float* b_out = (const float*)d_in[5];
    const float* res_k = (const float*)d_in[6];
    float* out = (float*)d_out;

    // ---- workspace carve-up (aliased; ~480 MB total) ----
    unsigned char* wp = (unsigned char*)d_ws;
    auto take = [&](size_t bytes) -> void* {
        void* p = wp; wp += (bytes + 255) & ~(size_t)255; return p;
    };
    __bf16* xn     = (__bf16*)take((size_t)BN * DD * 2);        // later aliased: concat
    __bf16* wqkvb  = (__bf16*)take((size_t)DD * 1536 * 2);
    __bf16* woutb  = (__bf16*)take((size_t)DD * DD * 2);
    __bf16* qb     = (__bf16*)take((size_t)BH * NN * DH * 2);
    __bf16* kb     = (__bf16*)take((size_t)BH * NN * DH * 2);
    __bf16* vb     = (__bf16*)take((size_t)BH * NN * DH * 2);
    __bf16* qlb    = (__bf16*)take((size_t)BH * MM * DH * 2);
    __bf16* klb    = (__bf16*)take((size_t)BH * MM * DH * 2);
    void*   big1   = take((size_t)BH * NN * MM * 2);            // qkv_tmp -> attn1
    void*   big2   = take((size_t)BH * MM * NN * 2);            // attn3 -> outh(f32)
    float*  attn2f = (float*)take((size_t)BH * MM * MM * 4);
    __bf16* a_bf   = (__bf16*)take((size_t)BH * MM * MM * 2);
    __bf16* zA     = (__bf16*)take((size_t)BH * MM * MM * 2);
    __bf16* zB     = (__bf16*)take((size_t)BH * MM * MM * 2);
    __bf16* azb    = (__bf16*)take((size_t)BH * MM * MM * 2);
    __bf16* tA     = (__bf16*)take((size_t)BH * MM * MM * 2);
    __bf16* tB     = (__bf16*)take((size_t)BH * MM * MM * 2);
    float*  nb     = (float*)take(1024);
    __bf16* w1b    = (__bf16*)take((size_t)BH * MM * DH * 2);
    __bf16* w2b    = (__bf16*)take((size_t)BH * MM * DH * 2);

    __bf16* qkvtmp = (__bf16*)big1;   // [BN,1536]
    __bf16* attn1  = (__bf16*)big1;   // [bh,4096,256]
    __bf16* attn3  = (__bf16*)big2;   // [bh,256,4096]
    float*  outh   = (float*)big2;    // [bh,4096,64] f32 (attn3 dead by then)
    __bf16* concat = xn;              // [b,n,512]    (xn dead after QKV GEMM)

    const long sQ  = (long)NN * DH;        // per-bh q/k/v stride
    const long sL  = (long)MM * DH;        // per-bh landmark stride
    const long sM2 = (long)MM * MM;        // per-bh 256x256 stride
    const long s13 = (long)NN * MM;        // per-bh attn1/attn3 stride

    // 1) LayerNorm + weight conversion
    k_layernorm<<<BN, 256, 0, stream>>>(x, gamma, beta, xn);
    k_cvt_bf16<<<(DD * 1536) / 256, 256, 0, stream>>>(w_qkv, wqkvb, DD * 1536);
    k_cvt_bf16<<<(DD * DD) / 256, 256, 0, stream>>>(w_out, woutb, DD * DD);

    // 2) QKV projection: [32768,512] x [512,1536] -> bf16
    k_gemm<false, true, false><<<dim3(1536 / 64, BN / 64, 1), 128, 0, stream>>>(
        xn, wqkvb, qkvtmp, DD, DD, 1536, 1536, 0, 0, 0, 1.f, 0.f, nullptr, nullptr);
    k_split_qkv<<<(size_t)BN * DD / 256, 256, 0, stream>>>(qkvtmp, qb, kb, vb);

    // 3) landmarks
    k_landmark<<<(BH * MM * DH) / 256, 256, 0, stream>>>(qb, qlb);
    k_landmark<<<(BH * MM * DH) / 256, 256, 0, stream>>>(kb, klb);

    // 4) similarity GEMMs (Bt-mode: Bt = k_l / k row-major)
    k_gemm<true, true, false><<<dim3(MM / 64, NN / 64, BH), 128, 0, stream>>>(
        qb, klb, attn1, DH, DH, DH, MM, sQ, sL, s13, 1.f, 0.f, nullptr, nullptr);
    k_gemm<true, false, false><<<dim3(MM / 64, MM / 64, BH), 128, 0, stream>>>(
        qlb, klb, attn2f, DH, DH, DH, MM, sL, sL, sM2, 1.f, 0.f, nullptr, nullptr);
    k_gemm<true, true, false><<<dim3(NN / 64, MM / 64, BH), 128, 0, stream>>>(
        qlb, kb, attn3, DH, DH, DH, NN, sL, sQ, s13, 1.f, 0.f, nullptr, nullptr);

    // 5) softmaxes (in-place)
    k_softmax_bf16<<<(BH * NN) / 8, 256, 0, stream>>>(attn1, MM);
    k_softmax_bf16<<<(BH * MM) / 8, 256, 0, stream>>>(attn3, NN);
    k_softmax_f32_dual<<<(BH * MM) / 8, 256, 0, stream>>>(attn2f, a_bf);

    // 6) Moore-Penrose pinv of attn2 (Newton-Schulz, 6 iters, bf16 WMMA)
    k_pinv_norm<<<BH, 256, 0, stream>>>(attn2f, nb);
    k_pinv_scalar<<<1, 64, 0, stream>>>(nb);
    k_z0<<<(BH * MM * MM) / 256, 256, 0, stream>>>(attn2f, nb, zA);
    __bf16* zc = zA; __bf16* zn = zB;
    for (int it = 0; it < 6; ++it) {
        // az = a @ z
        k_gemm<false, true, false><<<dim3(4, 4, BH), 128, 0, stream>>>(
            a_bf, zc, azb, MM, MM, MM, MM, sM2, sM2, sM2, 1.f, 0.f, nullptr, nullptr);
        // tA = 7I - az
        k_diag7<<<(BH * MM * MM) / 256, 256, 0, stream>>>(azb, tA);
        // tB = 15I - az @ tA
        k_gemm<false, true, false><<<dim3(4, 4, BH), 128, 0, stream>>>(
            azb, tA, tB, MM, MM, MM, MM, sM2, sM2, sM2, -1.f, 15.f, nullptr, nullptr);
        // tA = 13I - az @ tB
        k_gemm<false, true, false><<<dim3(4, 4, BH), 128, 0, stream>>>(
            azb, tB, tA, MM, MM, MM, MM, sM2, sM2, sM2, -1.f, 13.f, nullptr, nullptr);
        // zn = 0.25 * zc @ tA
        k_gemm<false, true, false><<<dim3(4, 4, BH), 128, 0, stream>>>(
            zc, tA, zn, MM, MM, MM, MM, sM2, sM2, sM2, 0.25f, 0.f, nullptr, nullptr);
        __bf16* t = zc; zc = zn; zn = t;
    }

    // 7) out = attn1 @ (zc @ (attn3 @ v))   (associativity saves ~34 GFLOP)
    k_gemm<false, true, false><<<dim3(1, MM / 64, BH), 128, 0, stream>>>(
        attn3, vb, w1b, NN, NN, DH, DH, s13, sQ, sL, 1.f, 0.f, nullptr, nullptr);
    k_gemm<false, true, false><<<dim3(1, MM / 64, BH), 128, 0, stream>>>(
        zc, w1b, w2b, MM, MM, DH, DH, sM2, sL, sL, 1.f, 0.f, nullptr, nullptr);
    k_gemm<false, false, false><<<dim3(1, NN / 64, BH), 128, 0, stream>>>(
        attn1, w2b, outh, MM, MM, DH, DH, s13, sL, sQ, 1.f, 0.f, nullptr, nullptr);

    // 8) residual depthwise conv + head concat -> [b,n,512] bf16
    k_conv_combine<<<dim3(NN / 256, BH), 256, 0, stream>>>(vb, outh, res_k, concat);

    // 9) output projection + bias + input residual -> d_out f32
    k_gemm<false, false, true><<<dim3(DD / 64, BN / 64, 1), 128, 0, stream>>>(
        concat, woutb, out, DD, DD, DD, DD, 0, 0, 0, 1.f, 0.f, b_out, x);
}
